// TypeEmbedding_34943853920383
// MI455X (gfx1250) — compile-verified, matrix operand order checked by reading, not run on hardware
//
#include <hip/hip_runtime.h>
#include <stdint.h>

// Problem sizes (match reference)
#define B_SZ 16384
#define H_SZ 1024
#define E_SZ 64
#define D_SZ 128
#define ESPLIT 4
#define EPW (E_SZ / ESPLIT)

typedef __attribute__((ext_vector_type(16))) __bf16    v16bf;
typedef __attribute__((ext_vector_type(2)))  __bf16    v2bf;
typedef __attribute__((ext_vector_type(8)))  float     v8f;
typedef __attribute__((ext_vector_type(8)))  uint32_t  v8u;

__device__ __forceinline__ uint32_t pack2bf(float lo, float hi) {
  v2bf v = {(__bf16)lo, (__bf16)hi};                 // native v_cvt_pk_bf16_f32 path
  return __builtin_bit_cast(uint32_t, v);
}
__device__ __forceinline__ unsigned short f2bf16(float f) {
  return __builtin_bit_cast(unsigned short, (__bf16)f);
}
__device__ __forceinline__ v8f splat8(float x) {
  v8f v = {x, x, x, x, x, x, x, x};
  return v;
}
__device__ __forceinline__ v8f wmma_bf16(v16bf a, v16bf b, v8f c) {
  // D = A(16x32 bf16) * B(32x16 bf16) + C(16x16 f32)
  return __builtin_amdgcn_wmma_f32_16x16x32_bf16(false, a, false, b, (short)0, c,
                                                 false, false);
}
__device__ __forceinline__ v16bf load_bfrag(const uint32_t* p) {
  return __builtin_bit_cast(v16bf, *(const v8u*)p);
}

// ---------------------------------------------------------------------------
// Prep: convert We [E,128,128] f32 -> bf16 B-fragment layout
//   frag index = ((e*4 + ks)*8 + nt)  (ks = K-step of 32, nt = N-tile of 16)
//   per-lane 8 dwords (16 bf16): lane = n%16 + 16*(K-subgroup odd)
// ---------------------------------------------------------------------------
__global__ __launch_bounds__(256) void prep_we_kernel(const float* __restrict__ we,
                                                      unsigned short* __restrict__ dst) {
  int tid = blockIdx.x * 256 + threadIdx.x;          // e*16384 + k*128 + n
  int e = tid >> 14;
  int rem = tid & 16383;
  int k = rem >> 7;
  int n = rem & 127;
  int ks = k >> 5, kk = k & 31;
  int nt = n >> 4, nl = n & 15;
  int g = kk >> 3;                                   // K sub-group of 8
  int lane = nl + ((g & 1) << 4);
  int jj = ((kk & 7) >> 1) + ((g >> 1) << 2);        // dword 0..7
  int half = kk & 1;
  int idx = ((((e * 4 + ks) * 8 + nt) * 32 + lane) * 8 + jj) * 2 + half;
  dst[idx] = f2bf16(we[tid]);
}

// Same B-fragment layout for Wh [1024,128] (32 K-steps)
__global__ __launch_bounds__(256) void prep_wh_kernel(const float* __restrict__ wh,
                                                      unsigned short* __restrict__ dst) {
  int tid = blockIdx.x * 256 + threadIdx.x;          // k*128 + n
  int k = tid >> 7;
  int n = tid & 127;
  int ks = k >> 5, kk = k & 31;
  int nt = n >> 4, nl = n & 15;
  int g = kk >> 3;
  int lane = nl + ((g & 1) << 4);
  int jj = ((kk & 7) >> 1) + ((g >> 1) << 2);
  int half = kk & 1;
  int idx = (((ks * 8 + nt) * 32 + lane) * 8 + jj) * 2 + half;
  dst[idx] = f2bf16(wh[tid]);
}

// ---------------------------------------------------------------------------
// GEMM1: r = hidden @ Wh + Wh_b   (16384x128, K=1024)
// Writes r in bf16 A-fragment layout: rfrag[(btile*4 + ks)*32 + lane][8 dw]
// ---------------------------------------------------------------------------
__global__ __launch_bounds__(256) void gemm1_kernel(const float* __restrict__ hidden,
                                                    const float* __restrict__ wh_b,
                                                    const uint32_t* __restrict__ whfrag,
                                                    uint32_t* __restrict__ rfrag) {
  __shared__ unsigned short lds[8 * 16 * 128];       // 32 KB, per-wave 4 KB regions
  const int lane = threadIdx.x & 31;
  const int wave = threadIdx.x >> 5;
  const int m = lane & 15;
  const int hi = lane >> 4;
  const int btile = blockIdx.x * 8 + wave;
  const int rowbase = btile * 16;
  unsigned short* ldsw = lds + wave * (16 * 128);

  v8f acc[8];
#pragma unroll
  for (int nt = 0; nt < 8; ++nt) acc[nt] = splat8(0.0f);

  for (int ks = 0; ks < 32; ++ks) {
    // Stage 16 rows x 32 K of hidden into LDS as bf16 (stride 32, per-wave region)
    {
      const int r = lane >> 1;
      const int kc = (lane & 1) * 16;
      const float4* src =
          (const float4*)(hidden + (size_t)(rowbase + r) * H_SZ + ks * 32 + kc);
      uint32_t* dstp = (uint32_t*)(ldsw + r * 32 + kc);
#pragma unroll
      for (int t = 0; t < 4; ++t) {
        float4 v = src[t];
        dstp[t * 2 + 0] = pack2bf(v.x, v.y);
        dstp[t * 2 + 1] = pack2bf(v.z, v.w);
      }
    }
    asm volatile("s_wait_dscnt 0" ::: "memory");     // per-wave LDS RAW fence
    // Build A-fragment (16x32 bf16) for this wave's 16 rows
    v8u af;
#pragma unroll
    for (int jj = 0; jj < 8; ++jj) {
      int k = 2 * jj + (jj < 4 ? (hi ? 8 : 0) : (hi ? 16 : 8));
      af[jj] = *(const uint32_t*)(ldsw + m * 32 + k);
    }
    v16bf a = __builtin_bit_cast(v16bf, af);
#pragma unroll
    for (int nt = 0; nt < 8; ++nt) {
      v16bf b = load_bfrag(whfrag + ((ks * 8 + nt) * 32 + lane) * 8);
      acc[nt] = wmma_bf16(a, b, acc[nt]);
    }
  }

  // Bias, convert to bf16, and re-shape C-layout -> A-fragment layout via LDS
#pragma unroll
  for (int nt = 0; nt < 8; ++nt) {
    float bias = wh_b[nt * 16 + m];
#pragma unroll
    for (int j = 0; j < 8; ++j) {
      int row = j + 8 * hi;
      ldsw[row * 128 + nt * 16 + m] = f2bf16(acc[nt][j] + bias);
    }
  }
  asm volatile("s_wait_dscnt 0" ::: "memory");
#pragma unroll
  for (int ks = 0; ks < 4; ++ks) {
    v8u f;
#pragma unroll
    for (int jj = 0; jj < 8; ++jj) {
      int k = ks * 32 + 2 * jj + (jj < 4 ? (hi ? 8 : 0) : (hi ? 16 : 8));
      f[jj] = *(const uint32_t*)(ldsw + m * 128 + k);
    }
    *(v8u*)(rfrag + (((size_t)btile * 4 + ks) * 32 + lane) * 8) = f;
  }
}

// ---------------------------------------------------------------------------
// Fused: per entity e: logits = r@We[e]+b -> softmax -> outputs (store f32)
//        vt = outputs@We[e]+b, result += sum_e vt (atomicAdd, split over E)
// One wave owns a 16-row B-tile; A-fragments of r loaded once, reused EPW x.
// ---------------------------------------------------------------------------
__global__ __launch_bounds__(256) void fused_kernel(const uint32_t* __restrict__ rfrag,
                                                    const uint32_t* __restrict__ wefrag,
                                                    const float* __restrict__ we_b,
                                                    float* __restrict__ outputs,
                                                    float* __restrict__ result) {
  __shared__ unsigned short lds[8 * 16 * 128];       // 32 KB
  const int lane = threadIdx.x & 31;
  const int wave = threadIdx.x >> 5;
  const int m = lane & 15;
  const int hi = lane >> 4;
  const int btile = blockIdx.x * 8 + wave;
  const int rowbase = btile * 16;
  unsigned short* ldsw = lds + wave * (16 * 128);
  const int e0 = blockIdx.y * EPW;

  // Preload A-fragments of r for this 16-row tile (reused for every entity)
  v16bf ar[4];
#pragma unroll
  for (int ks = 0; ks < 4; ++ks)
    ar[ks] = load_bfrag(rfrag + (((size_t)btile * 4 + ks) * 32 + lane) * 8);

  v8f res[8];
#pragma unroll
  for (int nt = 0; nt < 8; ++nt) res[nt] = splat8(0.0f);

  for (int e = e0; e < e0 + EPW; ++e) {
    const uint32_t* wf = wefrag + (size_t)e * (4 * 8 * 32 * 8);

    // Prefetch next entity's 32KB fragment block (1KB stride per lane covers it)
    {
      int en = (e + 1 < E_SZ) ? e + 1 : e;
      const uint32_t* wfn = wefrag + (size_t)en * (4 * 8 * 32 * 8) + lane * 256;
      __builtin_prefetch(wfn, 0, 3);                 // global_prefetch_b8
    }

    // ---- GEMM2: logits ----
    v8f acc[8];
#pragma unroll
    for (int nt = 0; nt < 8; ++nt) acc[nt] = splat8(we_b[e * D_SZ + nt * 16 + m]);
#pragma unroll
    for (int ks = 0; ks < 4; ++ks) {
#pragma unroll
      for (int nt = 0; nt < 8; ++nt) {
        v16bf b = load_bfrag(wf + ((ks * 8 + nt) * 32 + lane) * 8);
        acc[nt] = wmma_bf16(ar[ks], b, acc[nt]);
      }
    }

    // ---- softmax over D within each row (row j+8*hi; 128 cols = 8 nt x 16 lanes)
#pragma unroll
    for (int j = 0; j < 8; ++j) {
      float mx = acc[0][j];
#pragma unroll
      for (int nt = 1; nt < 8; ++nt) mx = fmaxf(mx, acc[nt][j]);
      mx = fmaxf(mx, __shfl_xor(mx, 1));
      mx = fmaxf(mx, __shfl_xor(mx, 2));
      mx = fmaxf(mx, __shfl_xor(mx, 4));
      mx = fmaxf(mx, __shfl_xor(mx, 8));             // stays within 16-lane half
      float s = 0.0f;
#pragma unroll
      for (int nt = 0; nt < 8; ++nt) {
        float ex = __expf(acc[nt][j] - mx);
        acc[nt][j] = ex;
        s += ex;
      }
      s += __shfl_xor(s, 1);
      s += __shfl_xor(s, 2);
      s += __shfl_xor(s, 4);
      s += __shfl_xor(s, 8);
      float inv = __builtin_amdgcn_rcpf(s);
#pragma unroll
      for (int nt = 0; nt < 8; ++nt) acc[nt][j] *= inv;
    }

    // ---- store outputs (f32) + bf16 copy into LDS for layout conversion ----
#pragma unroll
    for (int nt = 0; nt < 8; ++nt) {
#pragma unroll
      for (int j = 0; j < 8; ++j) {
        int row = j + 8 * hi;
        int d = nt * 16 + m;
        float v = acc[nt][j];
        outputs[((size_t)(rowbase + row) * E_SZ + e) * D_SZ + d] = v;
        ldsw[row * 128 + d] = f2bf16(v);
      }
    }
    asm volatile("s_wait_dscnt 0" ::: "memory");

    // ---- GEMM3: vt = outputs @ We[e] + b (reuse acc as accumulator) ----
#pragma unroll
    for (int nt = 0; nt < 8; ++nt) acc[nt] = splat8(we_b[e * D_SZ + nt * 16 + m]);
#pragma unroll
    for (int ks = 0; ks < 4; ++ks) {
      v8u f;
#pragma unroll
      for (int jj = 0; jj < 8; ++jj) {
        int k = ks * 32 + 2 * jj + (jj < 4 ? (hi ? 8 : 0) : (hi ? 16 : 8));
        f[jj] = *(const uint32_t*)(ldsw + m * 128 + k);
      }
      v16bf a2 = __builtin_bit_cast(v16bf, f);
#pragma unroll
      for (int nt = 0; nt < 8; ++nt) {
        v16bf b = load_bfrag(wf + ((ks * 8 + nt) * 32 + lane) * 8);
        acc[nt] = wmma_bf16(a2, b, acc[nt]);
      }
    }
#pragma unroll
    for (int nt = 0; nt < 8; ++nt) res[nt] += acc[nt];
  }

  // ---- result += sum over this entity-range ----
#pragma unroll
  for (int nt = 0; nt < 8; ++nt) {
#pragma unroll
    for (int j = 0; j < 8; ++j) {
      int row = j + 8 * hi;
      atomicAdd(result + (size_t)(rowbase + row) * D_SZ + nt * 16 + m, res[nt][j]);
    }
  }
}

// ---------------------------------------------------------------------------
extern "C" void kernel_launch(void* const* d_in, const int* in_sizes, int n_in,
                              void* d_out, int out_size, void* d_ws, size_t ws_size,
                              hipStream_t stream) {
  const float* hidden = (const float*)d_in[0];   // [B,H]
  const float* wh_w   = (const float*)d_in[1];   // [H,D]
  const float* wh_b   = (const float*)d_in[2];   // [D]
  const float* we_w   = (const float*)d_in[3];   // [E,D,D]
  const float* we_b   = (const float*)d_in[4];   // [E,D]

  float* outputs = (float*)d_out;                              // [B,E,D]
  float* result  = outputs + (size_t)B_SZ * E_SZ * D_SZ;       // [B,D]

  // Workspace layout (dwords): We frags (2 MB) | Wh frags (256 KB) | r frags (4 MB)
  uint32_t* ws32   = (uint32_t*)d_ws;
  uint32_t* wefrag = ws32;
  uint32_t* whfrag = wefrag + (size_t)E_SZ * 4 * 8 * 32 * 8;   // 524288 dw
  uint32_t* rfrag  = whfrag + (size_t)32 * 8 * 32 * 8;         // +65536 dw

  prep_we_kernel<<<(E_SZ * D_SZ * D_SZ) / 256, 256, 0, stream>>>(
      we_w, (unsigned short*)wefrag);
  prep_wh_kernel<<<(H_SZ * D_SZ) / 256, 256, 0, stream>>>(
      wh_w, (unsigned short*)whfrag);
  gemm1_kernel<<<B_SZ / 128, 256, 0, stream>>>(hidden, wh_b, whfrag, rfrag);
  hipMemsetAsync(result, 0, (size_t)B_SZ * D_SZ * sizeof(float), stream);
  fused_kernel<<<dim3(B_SZ / 128, ESPLIT), 256, 0, stream>>>(
      rfrag, wefrag, we_b, outputs, result);
}
